// CausalSelfAttentionGQA_68685116998333
// MI455X (gfx1250) — matrix operand order repaired and checked by reading, not compiled, not run on hardware
//
#include <hip/hip_runtime.h>
#include <hip/hip_bf16.h>
#include <math.h>

// ---------------------------------------------------------------------------
// Problem constants (match reference)
// ---------------------------------------------------------------------------
#define Bk   2
#define Tk   2048
#define Ck   2048
#define Hk   16
#define KVk  4
#define HDk  128
#define REPk (Hk / KVk)

typedef __bf16 bf16;
typedef __attribute__((ext_vector_type(16))) __bf16 bf16x16;
typedef __attribute__((ext_vector_type(8)))  __bf16 bf16x8;
typedef __attribute__((ext_vector_type(8)))  float  v8f;

#define WMMA_BF16(a, b, c) \
  __builtin_amdgcn_wmma_f32_16x16x32_bf16(false, (a), false, (b), (short)0, (c), false, false)

// Column permutation within a 32-wide K chunk so a WMMA A-fragment is one
// contiguous 16-element run per lane: store order [0..7, 16..23, 8..15, 24..31].
// Group map: 0->0, 1->2, 2->1, 3->3.
__device__ __forceinline__ int kgrp_swz(int g) { return (g == 1) ? 2 : ((g == 2) ? 1 : g); }

// ---------------------------------------------------------------------------
// fp32 -> bf16 conversion (grid-stride)
// ---------------------------------------------------------------------------
__global__ __launch_bounds__(256) void cvt_f32_bf16(const float* __restrict__ s,
                                                    bf16* __restrict__ d, size_t n) {
  size_t i = (size_t)blockIdx.x * blockDim.x + threadIdx.x;
  size_t stride = (size_t)gridDim.x * blockDim.x;
  for (; i < n; i += stride) d[i] = (bf16)s[i];
}

// ---------------------------------------------------------------------------
// Tiled bf16 WMMA GEMM: C[M,N] = A[M,K] * B[K,N], fp32 accumulate.
// BM=128, BN=64, BK=32. 256 threads = 8 waves; wave w owns rows w*16..w*16+15
// and all 64 cols (4 accumulators). LDS tiles are stored A row-major, B in fragment order, so every
// WMMA fragment is a single contiguous 32-byte LDS read per lane. Global loads
// for K-step k+1 are register double-buffered: issued right after the barrier
// so HBM/L2 latency hides behind the WMMA burst.
// ---------------------------------------------------------------------------
template <typename OutT>
__global__ __launch_bounds__(256) void gemm_bf16_wmma(const bf16* __restrict__ A,
                                                      const bf16* __restrict__ B,
                                                      OutT* __restrict__ C,
                                                      int M, int N, int K) {
  __shared__ __align__(32) bf16 sA[128 * 32];      // row-major rows, natural k order
  __shared__ __align__(32) bf16 sB[4 * 32 * 16];   // [nt][lane][k] fragment order

  const int tid  = threadIdx.x;
  const int lane = tid & 31;
  const int wave = tid >> 5;
  const int bm   = blockIdx.y * 128;
  const int bn   = blockIdx.x * 64;

  const int m  = lane & 15;
  const int n  = lane & 15;
  const int fh = (lane >> 4) * 16;   // fragment half select (A and B)

  // Precomputed staging coordinates (constant across K loop).
  const int arow0 = (tid * 2) >> 2,     ag0 = (tid * 2) & 3;
  const int arow1 = (tid * 2 + 1) >> 2, ag1 = (tid * 2 + 1) & 3;
  const int brow  = tid >> 3;
  const int bcol8 = (tid & 7) * 8;
  const int bnt   = bcol8 >> 4;
  const int bhalf = (brow >= 16) ? 16 : 0;
  const int bi    = brow & 15;

  const bf16* ApA0 = &A[(size_t)(bm + arow0) * K + ag0 * 8];
  const bf16* ApA1 = &A[(size_t)(bm + arow1) * K + ag1 * 8];
  const bf16* BpB  = &B[(size_t)brow * N + bn + bcol8];

  v8f acc[4];
#pragma unroll
  for (int t = 0; t < 4; ++t) acc[t] = (v8f){0.f, 0.f, 0.f, 0.f, 0.f, 0.f, 0.f, 0.f};

  // Prologue: fetch first tile into registers.
  bf16x8 ra0 = *(const bf16x8*)ApA0;
  bf16x8 ra1 = *(const bf16x8*)ApA1;
  bf16x8 rb  = *(const bf16x8*)BpB;

  for (int k0 = 0; k0 < K; k0 += 32) {
    // ---- Commit staged registers to LDS (A row-major, B in fragment order).
    *(bf16x8*)&sA[arow0 * 32 + ag0 * 8] = ra0;
    *(bf16x8*)&sA[arow1 * 32 + ag1 * 8] = ra1;
#pragma unroll
    for (int e = 0; e < 8; ++e) {
      int n_in = (bcol8 + e) & 15;
      sB[bnt * 512 + (bhalf + n_in) * 16 + bi] = rb[e];
    }
    __syncthreads();

    // ---- Issue next tile's global loads early (overlap with WMMA burst).
    if (k0 + 32 < K) {
      ra0 = *(const bf16x8*)(ApA0 + k0 + 32);
      ra1 = *(const bf16x8*)(ApA1 + k0 + 32);
      rb  = *(const bf16x8*)(BpB + (size_t)(k0 + 32) * N);
    }

    // ---- Fragments: one contiguous 32B LDS read each.
    bf16x16 af = *(const bf16x16*)&sA[(wave * 16 + m) * 32 + fh];
#pragma unroll
    for (int nt = 0; nt < 4; ++nt) {
      bf16x16 bfr = *(const bf16x16*)&sB[nt * 512 + lane * 16];
      acc[nt] = WMMA_BF16(af, bfr, acc[nt]);
    }
    __syncthreads();
  }

  // C/D layout: lane holds col n; VGPR r holds row r + (lane>>4)*8.
  const int mh = (lane >> 4) * 8;
#pragma unroll
  for (int nt = 0; nt < 4; ++nt)
#pragma unroll
    for (int r = 0; r < 8; ++r) {
      int row = bm + wave * 16 + mh + r;
      int col = bn + nt * 16 + n;
      C[(size_t)row * N + col] = (OutT)acc[nt][r];
    }
}

// ---------------------------------------------------------------------------
// RoPE in-place on bf16 [B,T,nh,HD]; thread handles pair (d, d+64), d<64.
// ---------------------------------------------------------------------------
__global__ __launch_bounds__(256) void rope_bf16(bf16* __restrict__ qk,
                                                 const float* __restrict__ cosT,
                                                 const float* __restrict__ sinT,
                                                 int nh, int total) {
  int i = blockIdx.x * blockDim.x + threadIdx.x;
  if (i >= total) return;                   // total = B*T*nh*64
  int d    = i & 63;
  int row  = i >> 6;                        // (b*T + t)*nh + h
  int t    = (row / nh) % Tk;
  size_t base = (size_t)row * HDk;
  float c0 = cosT[t * HDk + d];
  float s0 = sinT[t * HDk + d];
  float c1 = cosT[t * HDk + d + 64];
  float s1 = sinT[t * HDk + d + 64];
  float a  = (float)qk[base + d];
  float b  = (float)qk[base + d + 64];
  qk[base + d]      = (bf16)(a * c0 - b * s0);
  qk[base + d + 64] = (bf16)(b * c1 + a * s1);
}

// ---------------------------------------------------------------------------
// V transpose: [B,T,KV,HD] -> Vt[B,KV,HD,T], 32x32 tiles via LDS.
// Makes the flash P*V B-fragment a single contiguous 32B global load.
// ---------------------------------------------------------------------------
__global__ __launch_bounds__(256) void transpose_v(const bf16* __restrict__ Vsrc,
                                                   bf16* __restrict__ Vt) {
  __shared__ bf16 tile[32][33];
  const int t0  = blockIdx.x * 32;
  const int d0  = blockIdx.y * 32;
  const int b   = blockIdx.z / KVk;
  const int kv  = blockIdx.z % KVk;
  const int tr  = threadIdx.x & 31;   // fast index
  const int tc  = threadIdx.x >> 5;   // 8 slow rows per pass
#pragma unroll
  for (int r = 0; r < 32; r += 8)
    tile[tc + r][tr] = Vsrc[(((size_t)(b * Tk + t0 + tc + r)) * KVk + kv) * HDk + d0 + tr];
  __syncthreads();
#pragma unroll
  for (int r = 0; r < 32; r += 8)
    Vt[(((size_t)(b * KVk + kv)) * HDk + d0 + tc + r) * Tk + t0 + tr] = tile[tr][tc + r];
}

// ---------------------------------------------------------------------------
// Flash attention, one wave (32 threads) per (b, h, 16-query tile).
// Online softmax via LDS; scores and P*V through bf16 WMMA. All K fragments
// for a key tile are preloaded into distinct registers so the 8 score WMMAs
// pipeline behind the loads instead of stalling on s_wait_loadcnt 0.
// ---------------------------------------------------------------------------
__global__ __launch_bounds__(32) void flash_attn_gqa(const bf16* __restrict__ Q,
                                                     const bf16* __restrict__ K,
                                                     const bf16* __restrict__ Vt,
                                                     bf16* __restrict__ Att) {
  const int lane = threadIdx.x;
  const int qt   = blockIdx.x;          // T/16 tiles
  const int h    = blockIdx.y;
  const int b    = blockIdx.z;
  const int q0   = qt * 16;
  const int kv   = h / REPk;

  __shared__ __align__(32) float s_s[16][33];   // padded score tile
  __shared__ __align__(32) bf16  s_p[16][32];   // probabilities, natural column order
  __shared__ float s_rmax[16], s_rsum[16], s_rs[16];

  const int m  = lane & 15;
  const int n  = lane & 15;
  const int kh = (lane >> 4) * 8;
  const int fh = (lane >> 4) * 16;
  const int mh = (lane >> 4) * 8;

  if (lane < 16) { s_rmax[lane] = -1e30f; s_rsum[lane] = 0.f; }

  // Q fragments for 4 d-chunks of 32 (once per block).
  bf16x16 aq[4];
  {
    const bf16* qrow = Q + ((size_t)((b * Tk + q0 + m)) * Hk + h) * HDk;
#pragma unroll
    for (int c = 0; c < 4; ++c) {
      bf16x8 lo = *(const bf16x8*)&qrow[c * 32 + fh];
      bf16x8 hi = *(const bf16x8*)&qrow[c * 32 + fh + 8];
#pragma unroll
      for (int i = 0; i < 8; ++i) { aq[c][i] = lo[i]; aq[c][8 + i] = hi[i]; }
    }
  }

  v8f o[8];
#pragma unroll
  for (int j = 0; j < 8; ++j) o[j] = (v8f){0.f, 0.f, 0.f, 0.f, 0.f, 0.f, 0.f, 0.f};

  const float scale = 0.08838834764831845f;   // 1/sqrt(128)
  const int   nkt   = (q0 + 15) / 32 + 1;     // causal key-tile count (32-wide)

  // Per-lane K row bases (two key halves), advanced by key tile.
  const bf16* Kr0 = &K[(((size_t)(b * Tk + n)) * KVk + kv) * HDk + fh];
  const bf16* Kr1 = &K[(((size_t)(b * Tk + 16 + n)) * KVk + kv) * HDk + fh];
  const size_t kstep = (size_t)32 * KVk * HDk;         // 32 keys forward
  const bf16* Vbase = &Vt[(((size_t)(b * KVk + kv)) * HDk + n) * Tk + fh];

  for (int kt = 0; kt < nkt; ++kt) {
    const int kbase = kt * 32;

    // Prefetch next key tile's K/V lines into L2 (global_prefetch_b8).
    if (kt + 1 < nkt) {
      __builtin_prefetch(Kr0 + (size_t)(kt + 1) * kstep, 0, 1);
      __builtin_prefetch(Kr1 + (size_t)(kt + 1) * kstep, 0, 1);
      __builtin_prefetch(Vbase + (kt + 1) * 32, 0, 1);
    }

    // ---- Preload all 8 K fragments (distinct registers), then 8 WMMAs.
    bf16x16 bk[8];
#pragma unroll
    for (int c = 0; c < 4; ++c) {
      bk[c]     = *(const bf16x16*)(Kr0 + (size_t)kt * kstep + c * 32);
      bk[4 + c] = *(const bf16x16*)(Kr1 + (size_t)kt * kstep + c * 32);
    }
    v8f s0 = (v8f){0.f, 0.f, 0.f, 0.f, 0.f, 0.f, 0.f, 0.f};
    v8f s1 = s0;
#pragma unroll
    for (int c = 0; c < 4; ++c) {
      s0 = WMMA_BF16(aq[c], bk[c], s0);
      s1 = WMMA_BF16(aq[c], bk[4 + c], s1);
    }

    // ---- Scatter masked+scaled scores to LDS (C-layout).
#pragma unroll
    for (int r = 0; r < 8; ++r) {
      int row = mh + r;
      int qi  = q0 + row;
      s_s[row][n]      = (kbase + n      <= qi) ? s0[r] * scale : -1e30f;
      s_s[row][16 + n] = (kbase + 16 + n <= qi) ? s1[r] * scale : -1e30f;
    }
    asm volatile("s_wait_dscnt 0x0" ::: "memory");

    // ---- Online softmax: lane L (<16) owns query row L. Probabilities are
    // written at natural columns so the A-fragment read is contiguous.
    if (lane < 16) {
      float tmax = -1e30f;
#pragma unroll
      for (int c = 0; c < 32; ++c) tmax = fmaxf(tmax, s_s[lane][c]);
      float om = s_rmax[lane];
      float nm = fmaxf(om, tmax);
      float rs = __expf(om - nm);
      float ps = 0.f;
#pragma unroll
      for (int c = 0; c < 32; ++c) {
        float p = __expf(s_s[lane][c] - nm);
        int g   = c >> 3;
        s_p[lane][c] = (bf16)p;
        ps += p;
      }
      s_rsum[lane] = s_rsum[lane] * rs + ps;
      s_rmax[lane] = nm;
      s_rs[lane]   = rs;
    }
    asm volatile("s_wait_dscnt 0x0" ::: "memory");

    // ---- Rescale running output by exp(old_max - new_max) per row.
#pragma unroll
    for (int r = 0; r < 8; ++r) {
      float rs = s_rs[mh + r];
#pragma unroll
      for (int j = 0; j < 8; ++j) o[j][r] *= rs;
    }

    // ---- P fragment: one contiguous 32B LDS read (natural columns).
    bf16x16 ap = *(const bf16x16*)&s_p[m][fh];

    // ---- O += P * V ; preload 8 V fragments, then 8 WMMAs (pipelined).
    bf16x16 bv[8];
#pragma unroll
    for (int j = 0; j < 8; ++j)
      bv[j] = *(const bf16x16*)(Vbase + (size_t)(j * 16) * Tk + kbase);
#pragma unroll
    for (int j = 0; j < 8; ++j) o[j] = WMMA_BF16(ap, bv[j], o[j]);
  }

  asm volatile("s_wait_dscnt 0x0" ::: "memory");
  // ---- Normalize and store.
#pragma unroll
  for (int r = 0; r < 8; ++r) {
    float inv = 1.f / s_rsum[mh + r];
    size_t base = ((size_t)(b * Tk + q0 + mh + r) * Hk + h) * HDk;
#pragma unroll
    for (int j = 0; j < 8; ++j) Att[base + j * 16 + n] = (bf16)(o[j][r] * inv);
  }
}

// ---------------------------------------------------------------------------
// Host-side orchestration.
// ---------------------------------------------------------------------------
extern "C" void kernel_launch(void* const* d_in, const int* in_sizes, int n_in,
                              void* d_out, int out_size, void* d_ws, size_t ws_size,
                              hipStream_t stream) {
  (void)in_sizes; (void)n_in; (void)out_size; (void)ws_size;

  const float* x    = (const float*)d_in[0];
  const float* cosT = (const float*)d_in[1];
  const float* sinT = (const float*)d_in[2];
  const float* Wq   = (const float*)d_in[3];
  const float* Wk   = (const float*)d_in[4];
  const float* Wv   = (const float*)d_in[5];
  const float* Wo   = (const float*)d_in[6];
  float* out = (float*)d_out;

  const size_t NX   = (size_t)Bk * Tk * Ck;          // 8,388,608
  const size_t NWQ  = (size_t)Ck * Hk * HDk;         // 4,194,304
  const size_t NWKV = (size_t)Ck * KVk * HDk;        // 1,048,576
  const size_t NQ   = (size_t)Bk * Tk * Hk * HDk;    // 8,388,608
  const size_t NKV  = (size_t)Bk * Tk * KVk * HDk;   // 2,097,152

  char* ws = (char*)d_ws;
  size_t off = 0;
  auto alloc = [&](size_t elems) {
    bf16* p = (bf16*)(ws + off);
    off += ((elems * sizeof(bf16) + 255) & ~(size_t)255);
    return p;
  };
  bf16* xb  = alloc(NX);
  bf16* Wqb = alloc(NWQ);
  bf16* Wkb = alloc(NWKV);
  bf16* Wvb = alloc(NWKV);
  bf16* Wob = alloc(NWQ);
  bf16* Qb  = alloc(NQ);
  bf16* Kb  = alloc(NKV);
  bf16* Vb  = alloc(NKV);
  bf16* Vtb = alloc(NKV);
  bf16* Atb = alloc(NQ);

  // 1) Convert inputs to bf16.
  cvt_f32_bf16<<<4096, 256, 0, stream>>>(x,  xb,  NX);
  cvt_f32_bf16<<<2048, 256, 0, stream>>>(Wq, Wqb, NWQ);
  cvt_f32_bf16<<<1024, 256, 0, stream>>>(Wk, Wkb, NWKV);
  cvt_f32_bf16<<<1024, 256, 0, stream>>>(Wv, Wvb, NWKV);
  cvt_f32_bf16<<<2048, 256, 0, stream>>>(Wo, Wob, NWQ);

  const int M = Bk * Tk;   // 4096
  // 2) QKV projections (bf16 out).
  gemm_bf16_wmma<bf16><<<dim3(Hk * HDk / 64, M / 128), 256, 0, stream>>>(
      xb, Wqb, Qb, M, Hk * HDk, Ck);
  gemm_bf16_wmma<bf16><<<dim3(KVk * HDk / 64, M / 128), 256, 0, stream>>>(
      xb, Wkb, Kb, M, KVk * HDk, Ck);
  gemm_bf16_wmma<bf16><<<dim3(KVk * HDk / 64, M / 128), 256, 0, stream>>>(
      xb, Wvb, Vb, M, KVk * HDk, Ck);

  // 3) RoPE on Q and K (in place); transpose V for the flash P*V fragment.
  {
    int totQ = Bk * Tk * Hk * 64;
    int totK = Bk * Tk * KVk * 64;
    rope_bf16<<<(totQ + 255) / 256, 256, 0, stream>>>(Qb, cosT, sinT, Hk,  totQ);
    rope_bf16<<<(totK + 255) / 256, 256, 0, stream>>>(Kb, cosT, sinT, KVk, totK);
  }
  transpose_v<<<dim3(Tk / 32, HDk / 32, Bk * KVk), 256, 0, stream>>>(Vb, Vtb);

  // 4) Flash attention.
  flash_attn_gqa<<<dim3(Tk / 16, Hk, Bk), 32, 0, stream>>>(Qb, Kb, Vtb, Atb);

  // 5) Output projection (fp32 out).
  gemm_bf16_wmma<float><<<dim3(Ck / 64, M / 128), 256, 0, stream>>>(
      Atb, Wob, out, M, Ck, Ck);
}
